// SurfSageAutoencoder_58617713656484
// MI455X (gfx1250) — matrix operand order repaired and checked by reading, not compiled, not run on hardware
//
#include <hip/hip_runtime.h>
#include <hip/hip_bf16.h>
#include <math.h>

typedef __attribute__((ext_vector_type(16))) _Float16 v16h;
typedef __attribute__((ext_vector_type(8)))  float    v8f;

// ---------------------------------------------------------------------------
// Fragment loaders (wave32 WMMA 16x16x32 f16)
// A (16x32, MxK): lanes 0-15 row M=lane, lanes 16-31 row M=lane-16.
//   elements 0..7  = K (k0 + hi*8 .. +7), elements 8..15 = K (k0+16+hi*8 ..)
// B (32x16, KxN): lane holds column N=lane&15; same K order as A:
//   elements 0..7 = K (k0 + hi*8 ..), elements 8..15 = K (k0+16+hi*8 ..)
// ---------------------------------------------------------------------------
__device__ __forceinline__ v16h load_fragA(const _Float16* rowPtr, int k0, int hi) {
  union { v16h v; uint4 u[2]; } f;
  f.u[0] = *(const uint4*)(rowPtr + k0 + hi * 8);
  f.u[1] = *(const uint4*)(rowPtr + k0 + 16 + hi * 8);
  return f.v;
}

__device__ __forceinline__ v16h load_fragB(const _Float16* rowPtr, int k0, int hi) {
  union { v16h v; uint4 u[2]; } f;
  const _Float16* p = rowPtr + k0 + hi * 8;
  f.u[0] = *(const uint4*)(p);
  f.u[1] = *(const uint4*)(p + 16);
  return f.v;
}

// ---------------------------------------------------------------------------
// Small utility kernels
// ---------------------------------------------------------------------------
__global__ void cvt_f32_to_f16(const float* __restrict__ in, _Float16* __restrict__ out, int n) {
  int i = blockIdx.x * blockDim.x + threadIdx.x;
  if (i < n) out[i] = (_Float16)in[i];
}

__global__ void degree_kernel(const int* __restrict__ dst, float* __restrict__ deg, int E) {
  int i = blockIdx.x * blockDim.x + threadIdx.x;
  if (i < E)
    __hip_atomic_fetch_add(&deg[dst[i]], 1.0f, __ATOMIC_RELAXED, __HIP_MEMORY_SCOPE_AGENT);
}

__global__ void invdeg_kernel(float* __restrict__ deg, int N) {
  int i = blockIdx.x * blockDim.x + threadIdx.x;
  if (i < N) deg[i] = 1.0f / fmaxf(deg[i], 1.0f);
}

// one thread = 4 features of one edge (128-wide features, 32 threads/edge)
__global__ void scatter_kernel(const _Float16* __restrict__ h, const int* __restrict__ src,
                               const int* __restrict__ dst, float* __restrict__ agg, int total) {
  int t = blockIdx.x * blockDim.x + threadIdx.x;
  if (t >= total) return;
  int e = t >> 5;
  int c = (t & 31) << 2;
  const _Float16* p = h + (size_t)src[e] * 128 + c;
  float* q = agg + (size_t)dst[e] * 128 + c;
  union { uint2 u; _Float16 hv[4]; } v;
  v.u = *(const uint2*)p;
#pragma unroll
  for (int i = 0; i < 4; ++i)
    __hip_atomic_fetch_add(q + i, (float)v.hv[i], __ATOMIC_RELAXED, __HIP_MEMORY_SCOPE_AGENT);
}

// aggh = f16(agg * invdeg[row]) , feature width fixed at 128
__global__ void normalize_kernel(const float* __restrict__ agg, const float* __restrict__ invdeg,
                                 _Float16* __restrict__ aggh, int n) {
  int i = blockIdx.x * blockDim.x + threadIdx.x;
  if (i < n) aggh[i] = (_Float16)(agg[i] * invdeg[i >> 7]);
}

// ---------------------------------------------------------------------------
// Fused SAGE GEMM: out = act( A1*B1^T + A2*B2^T + bias )
// One wave per 16x16 output tile. A2 may be null (regressor path).
// Weight matrices are row-major [Nout, K] (so B fragment row = weight row).
// ---------------------------------------------------------------------------
__global__ void __launch_bounds__(256) sage_gemm_wmma(
    const _Float16* __restrict__ A1, const _Float16* __restrict__ B1,
    const _Float16* __restrict__ A2, const _Float16* __restrict__ B2,
    const float* __restrict__ bias, _Float16* __restrict__ outH,
    int M, int K, int Nout, int applyRelu) {
  const int lane = threadIdx.x & 31;
  const int wv   = threadIdx.x >> 5;
  const int lo = lane & 15, hi = lane >> 4;
  const int tilesN = Nout >> 4;
  const int tilesM = M >> 4;
  const int tile = blockIdx.x * 8 + wv;           // wave-uniform
  if (tile >= tilesM * tilesN) return;            // wave-uniform exit (EXEC stays all-1)
  const int tm = tile / tilesN, tn = tile % tilesN;
  const int m0 = tm << 4, n0 = tn << 4;

  const _Float16* a1row = A1 + (size_t)(m0 + lo) * K;
  const _Float16* b1row = B1 + (size_t)(n0 + lo) * K;
  v8f c = {};
  for (int k0 = 0; k0 < K; k0 += 32) {
    v16h a = load_fragA(a1row, k0, hi);
    v16h b = load_fragB(b1row, k0, hi);
    c = __builtin_amdgcn_wmma_f32_16x16x32_f16(false, a, false, b, (short)0, c, false, false);
  }
  if (A2 != nullptr) {
    const _Float16* a2row = A2 + (size_t)(m0 + lo) * K;
    const _Float16* b2row = B2 + (size_t)(n0 + lo) * K;
    for (int k0 = 0; k0 < K; k0 += 32) {
      v16h a = load_fragA(a2row, k0, hi);
      v16h b = load_fragB(b2row, k0, hi);
      c = __builtin_amdgcn_wmma_f32_16x16x32_f16(false, a, false, b, (short)0, c, false, false);
    }
  }
  const float bv = bias[n0 + lo];
#pragma unroll
  for (int r = 0; r < 8; ++r) {
    int m = m0 + hi * 8 + r;
    float v = c[r] + bv;
    if (applyRelu) v = fmaxf(v, 0.0f);
    outH[(size_t)m * Nout + n0 + lo] = (_Float16)v;
  }
}

// ---------------------------------------------------------------------------
// Fused decoder: block = 16 edges. hid = relu([lat[src],lat[dst]] @ W1^T + b1)
// built tile-by-tile into LDS by 8 waves, then 16 lanes reduce with w2 and
// apply sigmoid. h_pair is never materialized in HBM.
// ---------------------------------------------------------------------------
__global__ void __launch_bounds__(256) decoder_wmma(
    const _Float16* __restrict__ lat, const _Float16* __restrict__ W1,
    const float* __restrict__ b1, const float* __restrict__ w2,
    const float* __restrict__ b2, const int* __restrict__ src,
    const int* __restrict__ dst, float* __restrict__ out) {
  __shared__ float hid[16][132];
  const int lane = threadIdx.x & 31;
  const int wv = threadIdx.x >> 5;
  const int lo = lane & 15, hi = lane >> 4;
  const int m0 = blockIdx.x << 4;   // first edge of this block
  const int n0 = wv << 4;           // this wave's column tile

  const int e = m0 + lo;
  const _Float16* ps = lat + (size_t)src[e] * 64;
  const _Float16* pd = lat + (size_t)dst[e] * 64;
  const _Float16* brow = W1 + (size_t)(n0 + lo) * 128;

  v8f c = {};
  v16h a, b;
  a = load_fragA(ps, 0, hi);  b = load_fragB(brow, 0, hi);
  c = __builtin_amdgcn_wmma_f32_16x16x32_f16(false, a, false, b, (short)0, c, false, false);
  a = load_fragA(ps, 32, hi); b = load_fragB(brow, 32, hi);
  c = __builtin_amdgcn_wmma_f32_16x16x32_f16(false, a, false, b, (short)0, c, false, false);
  a = load_fragA(pd, 0, hi);  b = load_fragB(brow, 64, hi);
  c = __builtin_amdgcn_wmma_f32_16x16x32_f16(false, a, false, b, (short)0, c, false, false);
  a = load_fragA(pd, 32, hi); b = load_fragB(brow, 96, hi);
  c = __builtin_amdgcn_wmma_f32_16x16x32_f16(false, a, false, b, (short)0, c, false, false);

  const float bv = b1[n0 + lo];
#pragma unroll
  for (int r = 0; r < 8; ++r)
    hid[hi * 8 + r][n0 + lo] = fmaxf(c[r] + bv, 0.0f);

  __syncthreads();

  if (threadIdx.x < 16) {
    float acc = b2[0];
#pragma unroll 8
    for (int k = 0; k < 128; ++k) acc += hid[threadIdx.x][k] * w2[k];
    out[m0 + threadIdx.x] = 1.0f / (1.0f + __expf(-acc));
  }
}

// ---------------------------------------------------------------------------
// Host launcher
// ---------------------------------------------------------------------------
extern "C" void kernel_launch(void* const* d_in, const int* in_sizes, int n_in,
                              void* d_out, int out_size, void* d_ws, size_t ws_size,
                              hipStream_t stream) {
  (void)n_in; (void)out_size; (void)ws_size;

  const float* x      = (const float*)d_in[0];
  const int*   ei     = (const int*)d_in[1];
  const float* w_l0   = (const float*)d_in[2];
  const float* w_r0   = (const float*)d_in[3];
  const float* b0     = (const float*)d_in[4];
  const float* w_l1   = (const float*)d_in[5];
  const float* w_r1   = (const float*)d_in[6];
  const float* b1     = (const float*)d_in[7];
  const float* w_l2   = (const float*)d_in[8];
  const float* w_r2   = (const float*)d_in[9];
  const float* b2     = (const float*)d_in[10];
  const float* reg_w  = (const float*)d_in[11];
  const float* reg_b  = (const float*)d_in[12];
  const float* dec_w1 = (const float*)d_in[13];
  const float* dec_b1 = (const float*)d_in[14];
  const float* dec_w2 = (const float*)d_in[15];
  const float* dec_b2 = (const float*)d_in[16];

  const int N = in_sizes[0] / 128;
  const int E = in_sizes[1] / 2;
  const int* src = ei;
  const int* dst = ei + E;

  // ---- workspace carve-out (256B aligned bump allocator) ----
  char* p = (char*)d_ws;
  auto alloc = [&](size_t bytes) -> void* {
    void* r = (void*)p;
    p += (bytes + 255) & ~(size_t)255;
    return r;
  };
  float*    deg   = (float*)   alloc((size_t)N * 4);
  float*    agg   = (float*)   alloc((size_t)N * 128 * 4);
  _Float16* xh    = (_Float16*)alloc((size_t)N * 128 * 2);  // later reused as h2
  _Float16* aggh  = (_Float16*)alloc((size_t)N * 128 * 2);
  _Float16* h1h   = (_Float16*)alloc((size_t)N * 128 * 2);  // later reused as latent
  _Float16* h3h   = (_Float16*)alloc((size_t)N * 64 * 2);
  _Float16* wl0h  = (_Float16*)alloc(128 * 128 * 2);
  _Float16* wr0h  = (_Float16*)alloc(128 * 128 * 2);
  _Float16* wl1h  = (_Float16*)alloc(128 * 128 * 2);
  _Float16* wr1h  = (_Float16*)alloc(128 * 128 * 2);
  _Float16* wl2h  = (_Float16*)alloc(64 * 128 * 2);
  _Float16* wr2h  = (_Float16*)alloc(64 * 128 * 2);
  _Float16* regwh = (_Float16*)alloc(64 * 64 * 2);
  _Float16* dw1h  = (_Float16*)alloc(128 * 128 * 2);

  _Float16* h2h  = xh;   // buffer reuse (xh dead after conv0 GEMM)
  _Float16* lath = h1h;  // buffer reuse (h1 dead after conv1 GEMM)

  auto g = [](int n) { return (n + 255) / 256; };

  // ---- f32 -> f16 conversions ----
  cvt_f32_to_f16<<<g(N * 128), 256, 0, stream>>>(x, xh, N * 128);
  cvt_f32_to_f16<<<g(128 * 128), 256, 0, stream>>>(w_l0, wl0h, 128 * 128);
  cvt_f32_to_f16<<<g(128 * 128), 256, 0, stream>>>(w_r0, wr0h, 128 * 128);
  cvt_f32_to_f16<<<g(128 * 128), 256, 0, stream>>>(w_l1, wl1h, 128 * 128);
  cvt_f32_to_f16<<<g(128 * 128), 256, 0, stream>>>(w_r1, wr1h, 128 * 128);
  cvt_f32_to_f16<<<g(64 * 128), 256, 0, stream>>>(w_l2, wl2h, 64 * 128);
  cvt_f32_to_f16<<<g(64 * 128), 256, 0, stream>>>(w_r2, wr2h, 64 * 128);
  cvt_f32_to_f16<<<g(64 * 64), 256, 0, stream>>>(reg_w, regwh, 64 * 64);
  cvt_f32_to_f16<<<g(128 * 128), 256, 0, stream>>>(dec_w1, dw1h, 128 * 128);

  // ---- degrees (computed once, inverted in place) ----
  hipMemsetAsync(deg, 0, (size_t)N * 4, stream);
  degree_kernel<<<g(E), 256, 0, stream>>>(dst, deg, E);
  invdeg_kernel<<<g(N), 256, 0, stream>>>(deg, N);

  auto launch_gemm = [&](const _Float16* A1, const _Float16* B1,
                         const _Float16* A2, const _Float16* B2,
                         const float* bias, _Float16* outH,
                         int M, int K, int Nout, int relu) {
    int tiles = (M / 16) * (Nout / 16);
    int blocks = (tiles + 7) / 8;
    sage_gemm_wmma<<<blocks, 256, 0, stream>>>(A1, B1, A2, B2, bias, outH, M, K, Nout, relu);
  };

  // ---- conv0: relu(lin_l(mean agg of x) + lin_r(x) + b0) ----
  hipMemsetAsync(agg, 0, (size_t)N * 128 * 4, stream);
  scatter_kernel<<<g(E * 32), 256, 0, stream>>>(xh, src, dst, agg, E * 32);
  normalize_kernel<<<g(N * 128), 256, 0, stream>>>(agg, deg, aggh, N * 128);
  launch_gemm(aggh, wl0h, xh, wr0h, b0, h1h, N, 128, 128, 1);

  // ---- conv1 ----
  hipMemsetAsync(agg, 0, (size_t)N * 128 * 4, stream);
  scatter_kernel<<<g(E * 32), 256, 0, stream>>>(h1h, src, dst, agg, E * 32);
  normalize_kernel<<<g(N * 128), 256, 0, stream>>>(agg, deg, aggh, N * 128);
  launch_gemm(aggh, wl1h, h1h, wr1h, b1, h2h, N, 128, 128, 1);

  // ---- conv2 (no relu, Nout = 64) ----
  hipMemsetAsync(agg, 0, (size_t)N * 128 * 4, stream);
  scatter_kernel<<<g(E * 32), 256, 0, stream>>>(h2h, src, dst, agg, E * 32);
  normalize_kernel<<<g(N * 128), 256, 0, stream>>>(agg, deg, aggh, N * 128);
  launch_gemm(aggh, wl2h, h2h, wr2h, b2, h3h, N, 128, 64, 0);

  // ---- regressor: latent = h3 @ reg_w^T + reg_b ----
  launch_gemm(h3h, regwh, nullptr, nullptr, reg_b, lath, N, 64, 64, 0);

  // ---- fused decoder MLP + sigmoid ----
  decoder_wmma<<<E / 16, 256, 0, stream>>>(lath, dw1h, dec_b1, dec_w2, dec_b2,
                                           src, dst, (float*)d_out);
}
